// MoEWithSharedExpert_71536975283025
// MI455X (gfx1250) — matrix operand order repaired and checked by reading, not compiled, not run on hardware
//
#include <hip/hip_runtime.h>
#include <math.h>

// ---------------------------------------------------------------------------
// MoE with shared expert, top-1 routing.  D=512, H=2048, E=4, T=8192.
// d_out layout (floats): [0, T*D) out | [T*D] aux | [T*D+1, T*D+1+T*E) gate
// ---------------------------------------------------------------------------

#define D_ 512
#define H_ 2048
#define E_ 4
#define T_ 8192
#define TM 64            // token tile (rows per block)
#define XS_STR 528       // Xs row stride (elems), 1056B, %16==0
#define W1S_STR 136      // W1s row stride, K-chunk 128 + pad 8 (256B + 16B pad)
#define W2S_STR 40       // W2s row stride, K-chunk 32 + pad 8  (64B + 16B pad)
#define HS_STR 80        // Hs row stride, 64 + pad 16

typedef __attribute__((ext_vector_type(16))) __bf16        v16bf;
typedef __attribute__((ext_vector_type(8)))  float         v8f;
typedef __attribute__((ext_vector_type(4)))  unsigned int  uv4;
typedef __attribute__((ext_vector_type(2)))  unsigned int  uv2;
typedef __attribute__((ext_vector_type(4)))  float         fv4;
typedef __attribute__((ext_vector_type(8)))  int           iv8;
typedef __attribute__((ext_vector_type(4)))  int           iv4;

#if __has_builtin(__builtin_amdgcn_tensor_load_to_lds) && \
    __has_builtin(__builtin_amdgcn_s_wait_tensorcnt)
#define HAVE_TDM 1
#else
#define HAVE_TDM 0
#endif

union FragU { v16bf v; uv4 q[2]; };

__device__ __forceinline__ unsigned short f2bf(float f) {
  union { float f; unsigned u; } c; c.f = f;
  unsigned u = c.u;
  return (unsigned short)((u + 0x7FFFu + ((u >> 16) & 1u)) >> 16);
}

__device__ __forceinline__ float gelu_f(float v) {
  return 0.5f * v * (1.0f + erff(v * 0.70710678118654752f));
}

// A fragment: 16x32 bf16, row = m0 + (lane&15), K(j) = (j>>3)*16 + (lane>>4)*8 + (j&7)
__device__ __forceinline__ v16bf load_frag_A(const unsigned short* base, int m0,
                                             int stride, int k0, int lane) {
  const unsigned short* p = base + (m0 + (lane & 15)) * stride + k0 + ((lane >> 4) << 3);
  FragU f;
  f.q[0] = *(const uv4*)(p);
  f.q[1] = *(const uv4*)(p + 16);
  return f.v;
}

// B fragment from LDS tile stored transposed [N][K]:
// N = n0 + (lane&15), K(j) = (j>>3)*16 + (lane>>4)*8 + (j&7)
__device__ __forceinline__ v16bf load_frag_B(const unsigned short* baseT, int n0,
                                             int stride, int k0, int lane) {
  const unsigned short* p = baseT + (n0 + (lane & 15)) * stride + k0 + ((lane >> 4) << 3);
  FragU f;
  f.q[0] = *(const uv4*)(p);
  f.q[1] = *(const uv4*)(p + 16);
  return f.v;
}

#if HAVE_TDM
// ---------------------------------------------------------------------------
// TDM 2-D tile load: bf16 tile (tile_rows x tile_k) from a row-major matrix
// (row stride row_stride_elems) into LDS at lds_off with pad_amount DWORD
// padding after every pad_interval-coded chunk (ISA 8.3/8.4 descriptor).
// Issue from ONE wave; caller barriers around it.
// ---------------------------------------------------------------------------
__device__ __forceinline__ void tdm_load_2d(unsigned lds_off, const void* gptr,
                                            unsigned tile_k_elems, unsigned tile_rows,
                                            unsigned row_stride_elems,
                                            unsigned pad_interval_code,
                                            unsigned pad_amount_code) {
  unsigned long long ga = (unsigned long long)(size_t)gptr;
  uv4 g0;
  g0[0] = 1u;                                   // count=1 (valid user descriptor)
  g0[1] = lds_off;                              // lds_addr [63:32]
  g0[2] = (unsigned)ga;                         // global_addr[31:0]
  g0[3] = (unsigned)((ga >> 32) & 0x01FFFFFFull) | (2u << 30);  // addr[56:32]|type=2
  const unsigned td0 = 0x7FFFFFFFu, td1 = 0x7FFFFFFFu;  // never OOB
  iv8 g1;
  g1[0] = (int)((1u << 16) |                    // data_size = 2 bytes
                (1u << 20) |                    // pad_enable
                (pad_interval_code << 22) | (pad_amount_code << 25));
  g1[1] = (int)((td0 & 0xFFFFu) << 16);         // abar=0 | tensor_dim0 lo
  g1[2] = (int)(((td0 >> 16) & 0xFFFFu) | ((td1 & 0xFFFFu) << 16));
  g1[3] = (int)(((td1 >> 16) & 0xFFFFu) | (tile_k_elems << 16));   // tile_dim0
  g1[4] = (int)(tile_rows & 0xFFFFu);           // tile_dim1 (tile_dim2 = 0)
  g1[5] = (int)row_stride_elems;                // tensor_dim0_stride[31:0]
  g1[6] = 0;
  g1[7] = 0;
  iv4 gz = {0, 0, 0, 0};
  iv8 gz8 = {0, 0, 0, 0, 0, 0, 0, 0};
  __builtin_amdgcn_tensor_load_to_lds(g0, g1, gz, gz, gz8, 0);
}
#endif

// ---------------------------------------------------------------------------
// Kernel 1: zero small ws counters (cnt[4], cursor[4], poff[5])
// ---------------------------------------------------------------------------
__global__ void init_kernel(unsigned* smallw) {
  if (threadIdx.x < 16) smallw[threadIdx.x] = 0u;
}

// ---------------------------------------------------------------------------
// Kernel 2: router. One wave per token.
// ---------------------------------------------------------------------------
__global__ __launch_bounds__(256)
void router_kernel(const float* __restrict__ x, const float* __restrict__ gW,
                   const float* __restrict__ gb, float* __restrict__ gate_out,
                   int* __restrict__ idxw, float* __restrict__ confw,
                   unsigned* __restrict__ cnt) {
  const int lane = threadIdx.x & 31;
  const int t = blockIdx.x * 8 + (threadIdx.x >> 5);
  float a0 = 0.f, a1 = 0.f, a2 = 0.f, a3 = 0.f;
  for (int d = lane; d < D_; d += 32) {
    float xv = x[(size_t)t * D_ + d];
    fv4 g = *(const fv4*)(gW + (size_t)d * E_);
    a0 += xv * g[0]; a1 += xv * g[1]; a2 += xv * g[2]; a3 += xv * g[3];
  }
  #pragma unroll
  for (int m = 16; m >= 1; m >>= 1) {
    a0 += __shfl_xor(a0, m, 32);
    a1 += __shfl_xor(a1, m, 32);
    a2 += __shfl_xor(a2, m, 32);
    a3 += __shfl_xor(a3, m, 32);
  }
  if (lane == 0) {
    float l[E_] = { a0 + gb[0], a1 + gb[1], a2 + gb[2], a3 + gb[3] };
    float mx = l[0];
    #pragma unroll
    for (int e = 1; e < E_; e++) mx = fmaxf(mx, l[e]);
    float p[E_], s = 0.f;
    #pragma unroll
    for (int e = 0; e < E_; e++) { p[e] = expf(l[e] - mx); s += p[e]; }
    float inv = 1.0f / s;
    int imax = 0; float best = p[0] * inv;
    #pragma unroll
    for (int e = 0; e < E_; e++) {
      float ge = p[e] * inv;
      gate_out[(size_t)t * E_ + e] = ge;
      if (ge > best) { best = ge; imax = e; }
    }
    idxw[t] = imax;
    confw[t] = best;
    atomicAdd(&cnt[imax], 1u);
  }
}

// ---------------------------------------------------------------------------
// Kernel 3: aux loss (deterministic tree reduction) + padded expert offsets.
// ---------------------------------------------------------------------------
__global__ __launch_bounds__(256)
void aux_offsets_kernel(const float* __restrict__ gate, const unsigned* __restrict__ cnt,
                        unsigned* __restrict__ poff, float* __restrict__ auxout) {
  __shared__ float red[256][E_];
  float s[E_] = {0.f, 0.f, 0.f, 0.f};
  for (int t = threadIdx.x; t < T_; t += 256) {
    #pragma unroll
    for (int e = 0; e < E_; e++) s[e] += gate[(size_t)t * E_ + e];
  }
  #pragma unroll
  for (int e = 0; e < E_; e++) red[threadIdx.x][e] = s[e];
  __syncthreads();
  for (int off = 128; off >= 1; off >>= 1) {
    if ((int)threadIdx.x < off) {
      #pragma unroll
      for (int e = 0; e < E_; e++) red[threadIdx.x][e] += red[threadIdx.x + off][e];
    }
    __syncthreads();
  }
  if (threadIdx.x == 0) {
    float aux = 0.f;
    poff[0] = 0u;
    #pragma unroll
    for (int e = 0; e < E_; e++) {
      aux += (float)cnt[e] * red[0][e];
      poff[e + 1] = poff[e] + ((cnt[e] + (TM - 1)) & ~(unsigned)(TM - 1));
    }
    auxout[0] = aux * ((float)E_ * 0.01f) / ((float)T_ * (float)T_);
  }
}

// ---------------------------------------------------------------------------
// Kernel 4: scatter tokens into padded per-expert lists.
// ---------------------------------------------------------------------------
__global__ __launch_bounds__(256)
void scatter_kernel(const int* __restrict__ idxw, const unsigned* __restrict__ poff,
                    unsigned* __restrict__ cursor, int* __restrict__ perm) {
  int t = blockIdx.x * 256 + threadIdx.x;
  int e = idxw[t];
  unsigned p = atomicAdd(&cursor[e], 1u);
  perm[poff[e] + p] = t;
}

// ---------------------------------------------------------------------------
// Kernel 4b (TDM path only): transpose + f32->bf16.  out[c][r] = bf16(in[r][c])
// in: R x C f32 (slice z), out: C x R bf16 (slice z).
// ---------------------------------------------------------------------------
__global__ __launch_bounds__(256)
void transpose_bf16_kernel(const float* __restrict__ in, unsigned short* __restrict__ outp,
                           int R, int C) {
  __shared__ float tile[32][33];
  const float* inp = in + (size_t)blockIdx.z * R * C;
  unsigned short* op = outp + (size_t)blockIdx.z * R * C;
  int tx = threadIdx.x & 31, ty = threadIdx.x >> 5;   // 32 x 8
  int c0 = blockIdx.x * 32, r0 = blockIdx.y * 32;
  for (int i = ty; i < 32; i += 8)
    tile[i][tx] = inp[(size_t)(r0 + i) * C + c0 + tx];
  __syncthreads();
  for (int i = ty; i < 32; i += 8)
    op[(size_t)(c0 + i) * R + r0 + tx] = f2bf(tile[tx][i]);
}

// ---------------------------------------------------------------------------
// Kernel 5: fused 2-GEMM MLP (bf16 WMMA, f32 accumulate).
//   SHARED=true : dense over all tokens; rows weighted (1-conf)*sigmoid(w); out "="
//   SHARED=false: gathered top-1 expert tiles; out "+="
//   TDM=true    : W tiles DMA'd by the Tensor Data Mover from pre-transposed
//                 bf16 weights (k-contiguous), LDS padding done by the TDM.
// 512 threads = 16 waves; out tile 64x512; wave -> rows (w&3)*16, cols (w>>2)*128.
// ---------------------------------------------------------------------------
template <bool SHARED, bool TDM>
__global__ __launch_bounds__(512)
void mlp_kernel(const float* __restrict__ x,
                const float* __restrict__ W1g, const float* __restrict__ b1g,
                const float* __restrict__ W2g, const float* __restrict__ b2g,
                const unsigned short* __restrict__ W1tg,   // [ (E)*H, D ] bf16 (n-major)
                const unsigned short* __restrict__ W2tg,   // [ (E)*D, H ] bf16 (n-major)
                const float* __restrict__ confw, const int* __restrict__ perm,
                const unsigned* __restrict__ cntp, const unsigned* __restrict__ poffp,
                const float* __restrict__ swp, float* __restrict__ out) {
  __shared__ __align__(16) unsigned short Xs[TM * XS_STR];     // 67584 B
  __shared__ __align__(16) unsigned short W1s[64 * W1S_STR];   // 17408 B
  __shared__ __align__(16) unsigned short W2s[D_ * W2S_STR];   // 40960 B
  __shared__ __align__(16) unsigned short Hs[TM * HS_STR];     // 10240 B
  __shared__ float wrow[TM];
  __shared__ int   tok[TM];

  const int tid = threadIdx.x;
  const int lane = tid & 31;
  const int w = tid >> 5;
  const unsigned row0 = blockIdx.x * (unsigned)TM;

  int e = 0;
  if (!SHARED) {
    if (row0 >= poffp[E_]) return;   // padded tail tiles: uniform early exit
    e = (row0 >= poffp[3]) ? 3 : (row0 >= poffp[2]) ? 2 : (row0 >= poffp[1]) ? 1 : 0;
  }
  const float* W1p = SHARED ? W1g : W1g + (size_t)e * D_ * H_;
  const float* W2p = SHARED ? W2g : W2g + (size_t)e * H_ * D_;
  const float* b1p = SHARED ? b1g : b1g + (size_t)e * H_;
  const float* b2p = SHARED ? b2g : b2g + (size_t)e * D_;
  const unsigned short* W1tp = SHARED ? W1tg : W1tg + (size_t)e * D_ * H_;
  const unsigned short* W2tp = SHARED ? W2tg : W2tg + (size_t)e * H_ * D_;

  if (tid < TM) {
    if (SHARED) {
      int t = (int)row0 + tid;
      tok[tid] = t;
      float sg = 1.0f / (1.0f + expf(-swp[0]));
      wrow[tid] = (1.0f - confw[t]) * sg;
    } else {
      unsigned pos = row0 + (unsigned)tid;
      bool act = pos < poffp[e] + cntp[e];
      tok[tid] = act ? perm[pos] : 0;
      wrow[tid] = act ? 1.0f : 0.0f;
    }
  }
  __syncthreads();

  // Stage X tile -> LDS bf16 (rows via token gather).
  for (int c = tid; c < TM * (D_ / 4); c += 512) {
    int r = c >> 7, c4 = c & 127;
    fv4 v = *(const fv4*)(x + (size_t)tok[r] * D_ + c4 * 4);
    uv2 pk;
    pk[0] = (unsigned)f2bf(v[0]) | ((unsigned)f2bf(v[1]) << 16);
    pk[1] = (unsigned)f2bf(v[2]) | ((unsigned)f2bf(v[3]) << 16);
    *(uv2*)&Xs[r * XS_STR + c4 * 4] = pk;
  }

#if HAVE_TDM
  const unsigned ldsW1 = (unsigned)(unsigned long long)(size_t)&W1s[0];
  const unsigned ldsW2 = (unsigned)(unsigned long long)(size_t)&W2s[0];
#endif

  const int m1 = (w & 3) << 4;          // wave's 16-row tile
  const int n1 = (w >> 2) << 4;         // GEMM1 col tile within 64-wide H chunk
  const int c0 = (w >> 2) << 7;         // GEMM2 128-col slice
  const int hb = (lane >> 4) << 3;      // C-layout row offset (0 or 8)
  const int cl = lane & 15;

  v8f oacc[8];
  #pragma unroll
  for (int i = 0; i < 8; i++) oacc[i] = (v8f){};

  for (int hc = 0; hc < H_; hc += 64) {
    // ---- GEMM1: htile(64x64) = Xs(64x512) @ W1[:, hc:hc+64] ----
    v8f hacc = (v8f){};
    for (int kc = 0; kc < D_; kc += 128) {
      __syncthreads();
      if constexpr (TDM) {
#if HAVE_TDM
        // rows = 64 H-columns (n), 128 K elems each, pad 4 DW per 64 DW
        if (w == 0) {
          tdm_load_2d(ldsW1, W1tp + (size_t)hc * D_ + kc, 128u, 64u, (unsigned)D_,
                      5u /*64 DW*/, 3u /*4 DW*/);
          __builtin_amdgcn_s_wait_tensorcnt(0);
        }
#endif
      } else {
        // stage W1 chunk [kc..kc+128) x [hc..hc+64) transposed -> W1s[n][k]
        for (int c = tid; c < 2048; c += 512) {
          int i = c >> 4, j4 = c & 15;
          fv4 v = *(const fv4*)(W1p + (size_t)(kc + i) * H_ + hc + j4 * 4);
          int jb = j4 * 4;
          W1s[(jb + 0) * W1S_STR + i] = f2bf(v[0]);
          W1s[(jb + 1) * W1S_STR + i] = f2bf(v[1]);
          W1s[(jb + 2) * W1S_STR + i] = f2bf(v[2]);
          W1s[(jb + 3) * W1S_STR + i] = f2bf(v[3]);
        }
      }
      __syncthreads();
      #pragma unroll
      for (int k = 0; k < 128; k += 32) {
        v16bf a = load_frag_A(Xs, m1, XS_STR, kc + k, lane);
        v16bf b = load_frag_B(W1s, n1, W1S_STR, k, lane);
        hacc = __builtin_amdgcn_wmma_f32_16x16x32_bf16(false, a, false, b,
                                                       (short)0, hacc, false, false);
      }
    }
    // ---- bias + GELU + row weight -> Hs (bf16) ----
    float b1v = b1p[hc + n1 + cl];
    #pragma unroll
    for (int i = 0; i < 8; i++) {
      int r = m1 + hb + i;
      float g = gelu_f(hacc[i] + b1v) * wrow[r];
      Hs[r * HS_STR + n1 + cl] = f2bf(g);
    }
    // ---- GEMM2: out(64x512) += Hs(64x64) @ W2[hc:hc+64, :] ----
    for (int kc2 = 0; kc2 < 64; kc2 += 32) {
      __syncthreads();
      if constexpr (TDM) {
#if HAVE_TDM
        // rows = 512 D-columns (n), 32 K elems each, pad 4 DW per 16 DW
        if (w == 0) {
          tdm_load_2d(ldsW2, W2tp + hc + kc2, 32u, 512u, (unsigned)H_,
                      3u /*16 DW*/, 3u /*4 DW*/);
          __builtin_amdgcn_s_wait_tensorcnt(0);
        }
#endif
      } else {
        // stage W2 chunk [hc+kc2..+32) x [0..512) transposed -> W2s[n][k]
        for (int c = tid; c < 4096; c += 512) {
          int i = c >> 7, j4 = c & 127;
          fv4 v = *(const fv4*)(W2p + (size_t)(hc + kc2 + i) * D_ + j4 * 4);
          int jb = j4 * 4;
          W2s[(jb + 0) * W2S_STR + i] = f2bf(v[0]);
          W2s[(jb + 1) * W2S_STR + i] = f2bf(v[1]);
          W2s[(jb + 2) * W2S_STR + i] = f2bf(v[2]);
          W2s[(jb + 3) * W2S_STR + i] = f2bf(v[3]);
        }
      }
      __syncthreads();
      v16bf a = load_frag_A(Hs, m1, HS_STR, kc2, lane);
      #pragma unroll
      for (int ct = 0; ct < 8; ct++) {
        v16bf b = load_frag_B(W2s, c0 + ct * 16, W2S_STR, 0, lane);
        oacc[ct] = __builtin_amdgcn_wmma_f32_16x16x32_bf16(false, a, false, b,
                                                           (short)0, oacc[ct], false, false);
      }
    }
  }

  // ---- epilogue ----
  #pragma unroll
  for (int ct = 0; ct < 8; ct++) {
    int col = c0 + ct * 16 + cl;
    float bias = b2p[col];
    #pragma unroll
    for (int i = 0; i < 8; i++) {
      int r = m1 + hb + i;
      float val = oacc[ct][i];
      if (SHARED) {
        out[(size_t)tok[r] * D_ + col] = val + wrow[r] * bias;   // first writer
      } else if (wrow[r] != 0.0f) {
        float* o = &out[(size_t)tok[r] * D_ + col];
        *o = *o + val + bias;                                    // add routed expert
      }
    }
  }
}

// ---------------------------------------------------------------------------
// Host launcher
// ---------------------------------------------------------------------------
extern "C" void kernel_launch(void* const* d_in, const int* in_sizes, int n_in,
                              void* d_out, int out_size, void* d_ws, size_t ws_size,
                              hipStream_t stream) {
  const float* x   = (const float*)d_in[0];
  const float* gW  = (const float*)d_in[1];
  const float* gb  = (const float*)d_in[2];
  const float* W1  = (const float*)d_in[3];
  const float* b1  = (const float*)d_in[4];
  const float* W2  = (const float*)d_in[5];
  const float* b2  = (const float*)d_in[6];
  const float* sW1 = (const float*)d_in[7];
  const float* sb1 = (const float*)d_in[8];
  const float* sW2 = (const float*)d_in[9];
  const float* sb2 = (const float*)d_in[10];
  const float* swt = (const float*)d_in[11];

  float* out      = (float*)d_out;                     // [T, D]
  float* auxout   = out + (size_t)T_ * D_;             // scalar
  float* gate_out = auxout + 1;                        // [T, E]

  char* ws = (char*)d_ws;
  int*      idxw   = (int*)ws;                         // T ints
  float*    confw  = (float*)(ws + 32768);             // T floats
  int*      perm   = (int*)(ws + 65536);               // T + E*TM ints
  unsigned* cnt    = (unsigned*)(ws + 99328);          // [4]
  unsigned* cursor = cnt + 4;                          // [4]
  unsigned* poff   = cnt + 8;                          // [5]

  // bf16 pre-transposed weights for the TDM path
  unsigned short* W1T  = (unsigned short*)(ws + (1 << 20));    // E*H*D
  unsigned short* W2T  = W1T + (size_t)E_ * D_ * H_;           // E*D*H
  unsigned short* sW1T = W2T + (size_t)E_ * D_ * H_;           // H*D
  unsigned short* sW2T = sW1T + (size_t)D_ * H_;               // D*H
  const size_t NEED = (size_t)(1 << 20) + 2u * (size_t)E_ * D_ * H_ * 2u
                    + 2u * (size_t)D_ * H_ * 2u;               // ~22 MB

  init_kernel<<<1, 64, 0, stream>>>(cnt);
  router_kernel<<<T_ / 8, 256, 0, stream>>>(x, gW, gb, gate_out, idxw, confw, cnt);
  aux_offsets_kernel<<<1, 256, 0, stream>>>(gate_out, cnt, poff, auxout);
  scatter_kernel<<<T_ / 256, 256, 0, stream>>>(idxw, poff, cursor, perm);

#if HAVE_TDM
  if (ws_size >= NEED) {
    // prepass: f32 -> bf16, n-major (k-contiguous) layouts for the TDM
    transpose_bf16_kernel<<<dim3(64, 16, E_), 256, 0, stream>>>(W1, W1T, D_, H_);
    transpose_bf16_kernel<<<dim3(16, 64, E_), 256, 0, stream>>>(W2, W2T, H_, D_);
    transpose_bf16_kernel<<<dim3(64, 16, 1), 256, 0, stream>>>(sW1, sW1T, D_, H_);
    transpose_bf16_kernel<<<dim3(16, 64, 1), 256, 0, stream>>>(sW2, sW2T, H_, D_);
    mlp_kernel<true, true><<<T_ / TM, 512, 0, stream>>>(
        x, sW1, sb1, sW2, sb2, sW1T, sW2T, confw, perm, cnt, poff, swt, out);
    mlp_kernel<false, true><<<T_ / TM + E_, 512, 0, stream>>>(
        x, W1, b1, W2, b2, W1T, W2T, confw, perm, cnt, poff, swt, out);
    return;
  }
#endif
  // fallback: manual LDS staging with in-flight f32->bf16 conversion
  mlp_kernel<true, false><<<T_ / TM, 512, 0, stream>>>(
      x, sW1, sb1, sW2, sb2, nullptr, nullptr, confw, perm, cnt, poff, swt, out);
  mlp_kernel<false, false><<<T_ / TM + E_, 512, 0, stream>>>(
      x, W1, b1, W2, b2, nullptr, nullptr, confw, perm, cnt, poff, swt, out);
}
